// MSDeformAttn_63531156242632
// MI455X (gfx1250) — matrix operand order repaired and checked by reading, not compiled, not run on hardware
//
#include <hip/hip_runtime.h>
#include <hip/hip_bf16.h>
#include <math.h>

// ---------------------------------------------------------------------------
// MSDeformAttn (dual-branch) for gfx1250 / MI455X.
//  - All 7 GEMMs run on v_wmma_f32_16x16x32_f16 (wave32, 16x16 C tile / wave).
//  - Weight N-tile (16 cols x K) staged to LDS once per block via
//    GLOBAL_LOAD_ASYNC_TO_LDS_B128 (ASYNCcnt) -> 8x weight-traffic reduction,
//    inner loop reads B fragments with ds_load_b128.
//  - Activations pre-converted to f16 (A-fragment loads are 2x b128 global).
//  - Bilinear deformable sampling shares loc/attn via LDS per (n,q) block.
// Problem constants (from reference): N=2, Lq=Len_in=5440 -> M=10880 rows,
// D=384, 2U=256, heads=8, hd=48, levels=4, points=4.
// ---------------------------------------------------------------------------

typedef __attribute__((ext_vector_type(16))) _Float16 v16h;
typedef __attribute__((ext_vector_type(8)))  _Float16 v8h;
typedef __attribute__((ext_vector_type(8)))  float    v8f;

#define M_ROWS 10880   // N * Lq  (== N * Len_in)
#define LQ     5440
#define LEN_IN 5440
#define D_MOD  384
#define PADK   8       // LDS stride pad (halves): keeps b128 alignment, skews banks

// ---------------- elementwise f32 -> f16 convert ---------------------------
__global__ void cvt_f16_kernel(const float* __restrict__ src,
                               _Float16* __restrict__ dst, int n) {
    int i = blockIdx.x * blockDim.x + threadIdx.x;
    if (i < n) dst[i] = (_Float16)src[i];
}

// ------------- weight transpose+convert: W[KxN] f32 -> Wt[NxK] f16 ---------
__global__ void wt_cvt_kernel(const float* __restrict__ W,
                              _Float16* __restrict__ Wt, int K, int Nout) {
    int i = blockIdx.x * blockDim.x + threadIdx.x;
    if (i >= K * Nout) return;
    int k = i / Nout, nn = i - k * Nout;
    Wt[(size_t)nn * K + k] = (_Float16)W[i];
}

// ---------------- generic WMMA GEMM: C = A(f16,MxK) @ Wt(f16,NxK)^T + bias --
// Block = 256 threads = 8 waves. One N-tile (16 cols) per block, shared by all
// waves; wave w computes M-tile (blockIdx/ntn)*8 + w. The 16xK weight slice is
// staged into LDS once per block with async loads, then each wave's K-loop is:
//   2x global_load_b128 (A frag) + 2x ds_load_b128 (B frag) + 1x WMMA.
// 16-bit A 16x32 layout: lanes 0-15 rows M=0..15, K = {8g+0..7, 16+8g+0..7},
// g = lane>>4; B columns mirrored (K = k0 + {8g+0..7, 16+8g+0..7}, same order as A).
__global__ __launch_bounds__(256)
void gemm_f16_wmma(const _Float16* __restrict__ A, int lda,
                   const _Float16* __restrict__ Wt,        // [Nout x K]
                   const float* __restrict__ bias,
                   void* __restrict__ Cout, int ldc, int col_off, int c_is_half,
                   int M, int Nout, int K) {
    __shared__ _Float16 lb[16 * (384 + PADK)];   // max K = 384

    int ntn    = Nout >> 4;
    int tn     = blockIdx.x % ntn;
    int tmbase = (blockIdx.x / ntn) * 8;         // 8 M-tiles per block (exact)
    int tid    = threadIdx.x;
    int ldb    = K + PADK;                       // halves; K%8==0 -> 16B aligned

    // ---- stage B tile (16 cols x K halves) into LDS via async DMA ----
    {
        int cpk    = K >> 3;                     // 16B chunks per column
        int chunks = 16 * cpk;                   // 512 (K=256) / 768 (K=384)
        for (int c = tid; c < chunks; c += 256) {
            int col = c / cpk, wi = c - col * cpk;
            const _Float16* g = Wt + (size_t)(tn * 16 + col) * K + wi * 8;
            unsigned lofs = (unsigned)(uintptr_t)(lb + col * ldb + wi * 8);
            asm volatile("global_load_async_to_lds_b128 %0, %1, off"
                         :: "v"(lofs), "v"((unsigned long long)(uintptr_t)g)
                         : "memory");
        }
        asm volatile("s_wait_asynccnt 0x0" ::: "memory");
        __syncthreads();
    }

    int wave = tid >> 5;
    int lane = tid & 31;
    int g    = lane >> 4;      // K half-group select
    int r    = lane & 15;      // row (for A) / column (for B) within tile
    int tm   = tmbase + wave;

    const _Float16* arow = A + (size_t)(tm * 16 + r) * lda;
    const _Float16* brow = lb + r * ldb;         // LDS, column r of the N-tile

    v8f c = {};
    for (int k0 = 0; k0 < K; k0 += 32) {
        if (k0 + 64 <= K)                        // software prefetch next A chunk
            __builtin_prefetch(arow + k0 + 32, 0, 0);
        // A fragment: halves j=0..7 -> K=k0+8g+j ; j=8..15 -> K=k0+16+8g+(j-8)
        v8h alo = *(const v8h*)(arow + k0 + 8 * g);
        v8h ahi = *(const v8h*)(arow + k0 + 16 + 8 * g);
        // B fragment from LDS: halves j=0..7 -> K=k0+8g+j ; j=8..15 -> K=k0+16+8g+(j-8)
        v8h blo = *(const v8h*)(brow + k0 + 8 * g);
        v8h bhi = *(const v8h*)(brow + k0 + 16 + 8 * g);
        v16h a = __builtin_shufflevector(alo, ahi, 0,1,2,3,4,5,6,7,8,9,10,11,12,13,14,15);
        v16h b = __builtin_shufflevector(blo, bhi, 0,1,2,3,4,5,6,7,8,9,10,11,12,13,14,15);
        c = __builtin_amdgcn_wmma_f32_16x16x32_f16(
                /*neg_a=*/false, a, /*neg_b=*/false, b,
                /*c_mod=*/(short)0, c, /*reuse_a=*/false, /*reuse_b=*/false);
    }

    int   colg = tn * 16 + r;
    float bb   = bias ? bias[colg] : 0.0f;
    // C/D f32 layout: VGPR v -> row (v + 8g), column = lane&15
    if (c_is_half) {
        _Float16* C = (_Float16*)Cout;
        #pragma unroll
        for (int v = 0; v < 8; ++v) {
            int rowg = tm * 16 + v + 8 * g;
            C[(size_t)rowg * ldc + col_off + colg] = (_Float16)(c[v] + bb);
        }
    } else {
        float* C = (float*)Cout;
        #pragma unroll
        for (int v = 0; v < 8; ++v) {
            int rowg = tm * 16 + v + 8 * g;
            C[(size_t)rowg * ldc + col_off + colg] = c[v] + bb;
        }
    }
}

// ---------------- loc = ref + off / [W,H]  (in-place on off buffer) --------
__global__ void loc_kernel(float* __restrict__ loc,
                           const float* __restrict__ ref,
                           const int* __restrict__ ss, int total) {
    int i = blockIdx.x * blockDim.x + threadIdx.x;
    if (i >= total) return;
    int row = i >> 8;          // n*Lq + q
    int j   = i & 255;         // (h,l,p,c)
    int l   = (j >> 3) & 3;
    int cc  = j & 1;
    float norm = (float)(cc == 0 ? ss[l * 2 + 1] : ss[l * 2 + 0]); // [W,H]
    float rv   = ref[((size_t)row * 4 + l) * 2 + cc];
    loc[i] = rv + loc[i] / norm;
}

// ---------------- softmax over contiguous groups of 16 ---------------------
__global__ void softmax16_kernel(float* __restrict__ a, int ngroups) {
    int gidx = blockIdx.x * blockDim.x + threadIdx.x;
    if (gidx >= ngroups) return;
    float* p = a + (size_t)gidx * 16;
    float m = p[0];
    #pragma unroll
    for (int i = 1; i < 16; ++i) m = fmaxf(m, p[i]);
    float s = 0.0f;
    float e[16];
    #pragma unroll
    for (int i = 0; i < 16; ++i) { e[i] = __expf(p[i] - m); s += e[i]; }
    float inv = 1.0f / s;
    #pragma unroll
    for (int i = 0; i < 16; ++i) p[i] = e[i] * inv;
}

// ---------------- fused dual-branch deformable sampling ---------------------
// One 128-thread block per query row (n,q). Thread t: head h = t>>4,
// channels d0..d0+2 with d0 = (t&15)*3 (16*3 = 48 = hd). loc/attn in LDS.
__global__ __launch_bounds__(128)
void msda_sample_kernel(const float* __restrict__ loc,
                        const float* __restrict__ aw,
                        const float* __restrict__ awzd,
                        const _Float16* __restrict__ tv,   // (N*Len, 384) f16
                        const _Float16* __restrict__ bv,   // (N*Len, 384) f16
                        _Float16* __restrict__ tms,        // (M, 384) f16
                        _Float16* __restrict__ bms,        // (M, 384) f16
                        const int* __restrict__ ss,
                        const int* __restrict__ lsi) {
    __shared__ float sloc[256];
    __shared__ float sa[128];
    __shared__ float sb[128];

    int row = blockIdx.x;          // 0..M-1
    int n   = row / LQ;
    int t   = threadIdx.x;

    sloc[t]       = loc[(size_t)row * 256 + t];
    sloc[t + 128] = loc[(size_t)row * 256 + 128 + t];
    sa[t]         = aw  [(size_t)row * 128 + t];
    sb[t]         = awzd[(size_t)row * 128 + t];
    __syncthreads();

    int h  = t >> 4;
    int d0 = (t & 15) * 3;

    float at0 = 0.f, at1 = 0.f, at2 = 0.f;
    float ab0 = 0.f, ab1 = 0.f, ab2 = 0.f;

    #pragma unroll
    for (int l = 0; l < 4; ++l) {
        int H = ss[2 * l], W = ss[2 * l + 1];
        int start = lsi[l];
        size_t lvl_base = (size_t)(n * LEN_IN + start);
        #pragma unroll
        for (int p = 0; p < 4; ++p) {
            float x = sloc[h * 32 + l * 8 + p * 2 + 0] * (float)W - 0.5f;
            float y = sloc[h * 32 + l * 8 + p * 2 + 1] * (float)H - 0.5f;
            float xf = floorf(x), yf = floorf(y);
            float lx = x - xf, ly = y - yf;
            int x0 = (int)xf, y0 = (int)yf;
            float wa = sa[h * 16 + l * 4 + p];
            float wb = sb[h * 16 + l * 4 + p];
            float cw[4] = { (1.f - lx) * (1.f - ly), lx * (1.f - ly),
                            (1.f - lx) * ly,         lx * ly };
            int cx[4] = { x0, x0 + 1, x0,     x0 + 1 };
            int cy[4] = { y0, y0,     y0 + 1, y0 + 1 };
            #pragma unroll
            for (int cnr = 0; cnr < 4; ++cnr) {
                int xi = cx[cnr], yi = cy[cnr];
                if (xi >= 0 && xi < W && yi >= 0 && yi < H) {
                    size_t base = (lvl_base + (size_t)yi * W + xi) * 384
                                  + h * 48 + d0;
                    float wt = wa * cw[cnr];
                    float wz = wb * cw[cnr];
                    float v0 = (float)tv[base + 0];
                    float v1 = (float)tv[base + 1];
                    float v2 = (float)tv[base + 2];
                    at0 += wt * v0; at1 += wt * v1; at2 += wt * v2;
                    float u0 = (float)bv[base + 0];
                    float u1 = (float)bv[base + 1];
                    float u2 = (float)bv[base + 2];
                    ab0 += wz * u0; ab1 += wz * u1; ab2 += wz * u2;
                }
            }
        }
    }

    size_t o = (size_t)row * 384 + h * 48 + d0;
    tms[o + 0] = (_Float16)at0; tms[o + 1] = (_Float16)at1; tms[o + 2] = (_Float16)at2;
    bms[o + 0] = (_Float16)ab0; bms[o + 1] = (_Float16)ab1; bms[o + 2] = (_Float16)ab2;
}

// ---------------------------------------------------------------------------
extern "C" void kernel_launch(void* const* d_in, const int* in_sizes, int n_in,
                              void* d_out, int out_size, void* d_ws, size_t ws_size,
                              hipStream_t stream) {
    const float* q       = (const float*)d_in[0];   // (2,5440,384)
    const float* ref     = (const float*)d_in[1];   // (2,5440,4,2)
    const float* flat    = (const float*)d_in[2];   // (2,5440,384)
    const int*   ss      = (const int*)  d_in[3];   // (4,2)
    const int*   lsi     = (const int*)  d_in[4];   // (4,)
    const float* W_off     = (const float*)d_in[5];   // (256,256)
    const float* b_off     = (const float*)d_in[6];
    const float* W_attn    = (const float*)d_in[7];   // (256,128)
    const float* b_attn    = (const float*)d_in[8];
    const float* W_val     = (const float*)d_in[9];   // (256,384)
    const float* b_val     = (const float*)d_in[10];
    const float* W_out     = (const float*)d_in[11];  // (384,256)
    const float* b_out     = (const float*)d_in[12];
    const float* W_attn_zd = (const float*)d_in[13];  // (384,128)
    const float* b_attn_zd = (const float*)d_in[14];
    const float* W_val_zd  = (const float*)d_in[15];  // (384,384)
    const float* b_val_zd  = (const float*)d_in[16];
    const float* W_out_zd  = (const float*)d_in[17];  // (384,128)
    const float* b_out_zd  = (const float*)d_in[18];

    const int M = M_ROWS;

    // ---- workspace carve-up (all 256B aligned) ----
    char* w = (char*)d_ws;
    auto carve = [&](size_t bytes) -> char* {
        char* p = w;
        w += (bytes + 255) & ~(size_t)255;
        return p;
    };
    _Float16* qh    = (_Float16*)carve((size_t)M * 384 * 2);
    _Float16* fh    = (_Float16*)carve((size_t)M * 384 * 2);
    _Float16* Wt_off     = (_Float16*)carve(256 * 256 * 2);
    _Float16* Wt_attn    = (_Float16*)carve(128 * 256 * 2);
    _Float16* Wt_val     = (_Float16*)carve(384 * 256 * 2);
    _Float16* Wt_attn_zd = (_Float16*)carve(128 * 384 * 2);
    _Float16* Wt_val_zd  = (_Float16*)carve(384 * 384 * 2);
    _Float16* Wt_out     = (_Float16*)carve(256 * 384 * 2);
    _Float16* Wt_out_zd  = (_Float16*)carve(128 * 384 * 2);
    float*    locb  = (float*)carve((size_t)M * 256 * 4);
    float*    awb   = (float*)carve((size_t)M * 128 * 4);
    float*    awzdb = (float*)carve((size_t)M * 128 * 4);
    _Float16* tvh   = (_Float16*)carve((size_t)M * 384 * 2);
    _Float16* bvh   = (_Float16*)carve((size_t)M * 384 * 2);
    _Float16* tmsh  = (_Float16*)carve((size_t)M * 384 * 2);
    _Float16* bmsh  = (_Float16*)carve((size_t)M * 384 * 2);

    // ---- 1. convert activations to f16 ----
    {
        int n = M * 384;
        cvt_f16_kernel<<<(n + 255) / 256, 256, 0, stream>>>(q, qh, n);
        cvt_f16_kernel<<<(n + 255) / 256, 256, 0, stream>>>(flat, fh, n);
    }
    // ---- 2. transpose+convert weights ----
    auto wt = [&](const float* W, _Float16* Wt, int K, int Nout) {
        int n = K * Nout;
        wt_cvt_kernel<<<(n + 255) / 256, 256, 0, stream>>>(W, Wt, K, Nout);
    };
    wt(W_off,     Wt_off,     256, 256);
    wt(W_attn,    Wt_attn,    256, 128);
    wt(W_val,     Wt_val,     256, 384);
    wt(W_attn_zd, Wt_attn_zd, 384, 128);
    wt(W_val_zd,  Wt_val_zd,  384, 384);
    wt(W_out,     Wt_out,     384, 256);
    wt(W_out_zd,  Wt_out_zd,  384, 128);

    // ---- 3. WMMA GEMMs (projections) ----
    auto gemm = [&](const _Float16* A, int lda, const _Float16* Wt_,
                    const float* bias, void* C, int ldc, int coff,
                    int c_is_half, int Nout, int K) {
        int blocks = (M / 128) * (Nout / 16);   // exact: 85 * ntn
        gemm_f16_wmma<<<blocks, 256, 0, stream>>>(A, lda, Wt_, bias, C, ldc,
                                                  coff, c_is_half, M, Nout, K);
    };
    gemm(qh, 384, Wt_off,     b_off,     locb,  256, 0, 0, 256, 256); // offsets
    gemm(qh, 384, Wt_attn,    b_attn,    awb,   128, 0, 0, 128, 256); // attn logits
    gemm(qh, 384, Wt_attn_zd, b_attn_zd, awzdb, 128, 0, 0, 128, 384); // attn_zd
    gemm(fh, 384, Wt_val,     b_val,     tvh,   384, 0, 1, 384, 256); // top value
    gemm(fh, 384, Wt_val_zd,  b_val_zd,  bvh,   384, 0, 1, 384, 384); // bot value

    // ---- 4. loc transform + softmaxes ----
    {
        int total = M * 256;
        loc_kernel<<<(total + 255) / 256, 256, 0, stream>>>(locb, ref, ss, total);
        int ng = M * 8;
        softmax16_kernel<<<(ng + 255) / 256, 256, 0, stream>>>(awb, ng);
        softmax16_kernel<<<(ng + 255) / 256, 256, 0, stream>>>(awzdb, ng);
    }

    // ---- 5. fused dual-branch deformable sampling ----
    msda_sample_kernel<<<M, 128, 0, stream>>>(locb, awb, awzdb, tvh, bvh,
                                              tmsh, bmsh, ss, lsi);

    // ---- 6. output GEMMs straight into concatenated (.,384) output ----
    float* out = (float*)d_out;
    gemm(tmsh, 384, Wt_out,    b_out,    out, 384, 0,   0, 256, 384);
    gemm(bmsh, 384, Wt_out_zd, b_out_zd, out, 384, 256, 0, 128, 384);
}
